// TSCGraphAgent_3350074490946
// MI455X (gfx1250) — compile-verified, matrix operand order checked by reading, not run on hardware
//
#include <hip/hip_runtime.h>

// ---------------------------------------------------------------------------
// TSCGraphAgent forward for gfx1250 (MI455X).  All GEMM stages use
// v_wmma_f32_16x16x32_f16 (f16 inputs, f32 accumulate).  wave32.
// K1 additionally uses the Tensor Data Mover (tensor_load_to_lds) to DMA the
// hidden-state tile into LDS, fenced with s_wait_tensorcnt, and
// global_prefetch_b8 to pre-warm the L2-resident GRU weights.
// ---------------------------------------------------------------------------

typedef __attribute__((ext_vector_type(16))) _Float16 v16h;
typedef __attribute__((ext_vector_type(8)))  float    v8f;
typedef __attribute__((ext_vector_type(4)))  unsigned int v4u;
typedef __attribute__((ext_vector_type(8)))  unsigned int v8u;

#define BS       8
#define N_AGENTS 256
#define MAX_L    12
#define LDIM     27
#define HID      128
#define N_ACT    8
#define BN       (BS * N_AGENTS)        // 2048

// ---- f16 weight workspace layout (element offsets into half region) -------
#define OFF_FC1H 0                        // 128 x 32  (K padded 27->32)
#define OFF_WIH  4096                     // 384 x 128
#define OFF_WHH  53248                    // 384 x 128
#define OFF_GFC  102400                   // 128 x 128
#define OFF_GA1A 118784                   // 128 x 128  (ga1_w[:, :128])
#define OFF_GA1B 135168                   // 128 x 128  (ga1_w[:, 128:])
#define OFF_GOUT 151552                   // 128 x 128
#define OFF_FC2  167936                   // 16 x 256   (N padded 8->16)
#define HALF_TOT 172032                   // halfs
// f32 region follows (byte offset 344064, 256B aligned)
#define F_AGG 0                           // 2048 x 128
#define F_AI  262144                      // 2048 x 128
#define F_AJ  524288                      // 2048 x 128
// second half region (byte offset 344064 + 3145728)
#define H2_AGGH 0                         // 2048 x 128
#define H2_HGT  262144                    // 8 x 128 x 256  (hg transposed)
#define H2_AWH  524288                    // 8 x 256 x 256

// ---------------------------------------------------------------------------
// WMMA fragment helpers (ISA 7.12.2 layouts, wave32)
// ---------------------------------------------------------------------------
__device__ __forceinline__ v8f wmma_f16(v16h a, v16h b, v8f c) {
  return __builtin_amdgcn_wmma_f32_16x16x32_f16(
      /*neg_a=*/false, a, /*neg_b=*/false, b,
      /*c_mod=*/(short)0, c, /*reuse_a=*/false, /*reuse_b=*/false);
}

// A operand: row-major f16 matrix, row M = lane%16.
// halfs 0..7  = K[k0 + 8*(lane/16) .. +7]
// halfs 8..15 = K[k0 + 16 + 8*(lane/16) .. +7]
__device__ __forceinline__ v16h load_frag_a_h(const _Float16* base, int ld,
                                              int k0, int lane) {
  int m  = lane & 15;
  int ks = ((lane >> 4) & 1) * 8;
  const _Float16* p = base + m * ld + k0 + ks;
  v16h a;
#pragma unroll
  for (int j = 0; j < 8; ++j) { a[j] = p[j]; a[8 + j] = p[16 + j]; }
  return a;
}

// A operand sourced from f32 (LDS), converted to f16 in registers.
__device__ __forceinline__ v16h load_frag_a_f32(const float* base, int ld,
                                                int k0, int lane) {
  int m  = lane & 15;
  int ks = ((lane >> 4) & 1) * 8;
  const float* p = base + m * ld + k0 + ks;
  v16h a;
#pragma unroll
  for (int j = 0; j < 8; ++j) {
    a[j]     = (_Float16)p[j];
    a[8 + j] = (_Float16)p[16 + j];
  }
  return a;
}

// B operand: weight stored row-major [N][K]; column N = lane%16 (mirrors A).
__device__ __forceinline__ v16h load_frag_b_h(const _Float16* base, int ld,
                                              int n0, int k0, int lane) {
  int n  = lane & 15;
  int ks = ((lane >> 4) & 1) * 8;
  const _Float16* p = base + (n0 + n) * ld + k0 + ks;
  v16h b;
#pragma unroll
  for (int j = 0; j < 8; ++j) { b[j] = p[j]; b[8 + j] = p[16 + j]; }
  return b;
}

__device__ __forceinline__ float sigmoid_f(float x) {
  return 1.0f / (1.0f + __expf(-x));
}

// Issue a 1-D TDM load: nbytes contiguous bytes global -> LDS.
// data_size = 4B elements; nelem = nbytes/4.  Per-wave instruction (EXEC
// ignored); caller must s_wait_tensorcnt before consuming.
__device__ __forceinline__ void tdm_load_1d(unsigned lds_byte_addr,
                                            const void* gptr, unsigned nelem) {
  unsigned long long ga = (unsigned long long)gptr;
  v4u g0;
  g0.x = 1u;                                   // count=1, user mode, no gather
  g0.y = lds_byte_addr;                        // LDS destination (bytes)
  g0.z = (unsigned)ga;                         // global_addr[31:0]
  g0.w = (unsigned)((ga >> 32) & 0x1FFFFFFu)   // global_addr[56:32]
         | (2u << 30);                         // type = 2 ("image")
  v8u g1;
  g1.s0 = 0x20000u;                            // wg_mask=0, data_size=2 (4B)
  g1.s1 = (nelem & 0xFFFFu) << 16;             // tensor_dim0[15:0]
  g1.s2 = (nelem >> 16) & 0xFFFFu;             // tensor_dim0[31:16]
  g1.s3 = (nelem & 0xFFFFu) << 16;             // tile_dim0 (tile == tensor)
  g1.s4 = 0u;                                  // tile_dim1/2 = 0 (1-D)
  g1.s5 = nelem;                               // tensor_dim0_stride[31:0]
  g1.s6 = 0u;
  g1.s7 = 0u;
  asm volatile("tensor_load_to_lds %0, %1" :: "s"(g0), "s"(g1) : "memory");
}

// ---------------------------------------------------------------------------
// K0: pack weights to f16 with WMMA-legal padding
// ---------------------------------------------------------------------------
__global__ __launch_bounds__(256) void convert_weights_kernel(
    const float* __restrict__ fc1_w, const float* __restrict__ w_ih,
    const float* __restrict__ w_hh,  const float* __restrict__ gfc_w,
    const float* __restrict__ ga1_w, const float* __restrict__ gout_w,
    const float* __restrict__ fc2_w, _Float16* __restrict__ h) {
  int i = blockIdx.x * blockDim.x + threadIdx.x;
  if (i < 4096) {                                  // fc1: 128x27 -> 128x32
    int o = i >> 5, k = i & 31;
    h[OFF_FC1H + i] = (k < LDIM) ? (_Float16)fc1_w[o * LDIM + k] : (_Float16)0.f;
  } else if (i < 53248) {
    int j = i - 4096;  h[OFF_WIH + j] = (_Float16)w_ih[j];
  } else if (i < 102400) {
    int j = i - 53248; h[OFF_WHH + j] = (_Float16)w_hh[j];
  } else if (i < 118784) {
    int j = i - 102400; h[OFF_GFC + j] = (_Float16)gfc_w[j];
  } else if (i < 135168) {                         // ga1_w[:, :128]
    int j = i - 118784; int o = j >> 7, k = j & 127;
    h[OFF_GA1A + j] = (_Float16)ga1_w[o * 256 + k];
  } else if (i < 151552) {                         // ga1_w[:, 128:]
    int j = i - 135168; int o = j >> 7, k = j & 127;
    h[OFF_GA1B + j] = (_Float16)ga1_w[o * 256 + 128 + k];
  } else if (i < 167936) {
    int j = i - 151552; h[OFF_GOUT + j] = (_Float16)gout_w[j];
  } else if (i < HALF_TOT) {                       // fc2: 8x256 -> 16x256
    int j = i - 167936; int o = j >> 8, k = j & 255;
    h[OFF_FC2 + j] = (o < N_ACT) ? (_Float16)fc2_w[o * 256 + k] : (_Float16)0.f;
  }
}

// ---------------------------------------------------------------------------
// K1: lane encoder + GRU + lane attention, one agent per block (2048 blocks).
//     128 threads = 4 waves; every GEMM stage is WMMA on a padded 16-row tile.
//     Hidden tile DMA'd into LDS by the TDM; weights prefetched to cache.
// ---------------------------------------------------------------------------
__global__ __launch_bounds__(128) void lane_agent_kernel(
    const float* __restrict__ lf, const float* __restrict__ hin,
    const _Float16* __restrict__ fc1h, const float* __restrict__ fc1_b,
    const _Float16* __restrict__ wih,  const _Float16* __restrict__ whh,
    const float* __restrict__ b_ih, const float* __restrict__ b_hh,
    const float* __restrict__ lattn_w, const float* __restrict__ lattn_b,
    float* __restrict__ agg, _Float16* __restrict__ aggh) {
  __shared__ _Float16 s_lf[16 * 32];     // padded lane features (f16)
  __shared__ _Float16 s_x[16 * HID];     // relu(fc1) output (f16)
  __shared__ float    s_hin[16 * HID];   // previous hidden (f32)
  __shared__ float    s_h[16 * HID];     // GRU output (f32)
  __shared__ float    s_sc[16];
  __shared__ float    s_red[2];

  const int tid = threadIdx.x, lane = tid & 31, wave = tid >> 5;
  const int agent = blockIdx.x;

  // Pre-warm L2/WGP$ with the GRU weight matrices (shared by all 2048 blocks)
  __builtin_prefetch((const char*)wih + tid * 768, 0, 1);
  __builtin_prefetch((const char*)whh + tid * 768, 0, 1);

  // TDM: DMA the contiguous 12x128 f32 hidden tile straight into LDS.
  if (wave == 0) {
    tdm_load_1d((unsigned)(unsigned long long)(uintptr_t)&s_hin[0],
                hin + (size_t)agent * (MAX_L * HID), MAX_L * HID);
  }

  // stage lane features (zero padded rows 12..15, K 27..31) meanwhile
  for (int i = tid; i < 16 * 32; i += 128) {
    int m = i >> 5, k = i & 31;
    float v = (m < MAX_L && k < LDIM)
                  ? lf[(size_t)agent * (MAX_L * LDIM) + m * LDIM + k] : 0.f;
    s_lf[i] = (_Float16)v;
  }
  // zero the padded hidden rows 12..15 (disjoint from the DMA'd bytes)
  for (int i = tid; i < 4 * HID; i += 128) s_hin[MAX_L * HID + i] = 0.f;

  if (wave == 0) __builtin_amdgcn_s_wait_tensorcnt(0);
  __syncthreads();

  // x = relu(LF @ fc1^T + b1): 16x128, K=32 (one WMMA per 16-col tile)
  for (int t = wave; t < 8; t += 4) {
    v16h a = load_frag_a_h(s_lf, 32, 0, lane);
    v16h b = load_frag_b_h(fc1h, 32, t * 16, 0, lane);
    v8f c = {};
    c = wmma_f16(a, b, c);
    int col = t * 16 + (lane & 15), mh = ((lane >> 4) & 1) * 8;
    float bv = fc1_b[col];
#pragma unroll
    for (int r = 0; r < 8; ++r)
      s_x[(mh + r) * HID + col] = (_Float16)fmaxf(c[r] + bv, 0.f);
  }
  __syncthreads();

  // GRU: per H-tile keep 6 accumulators so the gate math fuses in registers
  for (int t = wave; t < 8; t += 4) {
    v8f giR = {}, giZ = {}, giN = {}, ghR = {}, ghZ = {}, ghN = {};
#pragma unroll
    for (int k0 = 0; k0 < HID; k0 += 32) {
      v16h ax = load_frag_a_h(s_x, HID, k0, lane);
      v16h ah = load_frag_a_f32(s_hin, HID, k0, lane);
      giR = wmma_f16(ax, load_frag_b_h(wih, HID,       t * 16, k0, lane), giR);
      giZ = wmma_f16(ax, load_frag_b_h(wih, HID, 128 + t * 16, k0, lane), giZ);
      giN = wmma_f16(ax, load_frag_b_h(wih, HID, 256 + t * 16, k0, lane), giN);
      ghR = wmma_f16(ah, load_frag_b_h(whh, HID,       t * 16, k0, lane), ghR);
      ghZ = wmma_f16(ah, load_frag_b_h(whh, HID, 128 + t * 16, k0, lane), ghZ);
      ghN = wmma_f16(ah, load_frag_b_h(whh, HID, 256 + t * 16, k0, lane), ghN);
    }
    int col = t * 16 + (lane & 15), mh = ((lane >> 4) & 1) * 8;
    float bir = b_ih[col], biz = b_ih[128 + col], bin = b_ih[256 + col];
    float bhr = b_hh[col], bhz = b_hh[128 + col], bhn = b_hh[256 + col];
#pragma unroll
    for (int r = 0; r < 8; ++r) {
      float rg = sigmoid_f(giR[r] + bir + ghR[r] + bhr);
      float zg = sigmoid_f(giZ[r] + biz + ghZ[r] + bhz);
      float ng = tanhf(giN[r] + bin + rg * (ghN[r] + bhn));
      float hp = s_hin[(mh + r) * HID + col];
      s_h[(mh + r) * HID + col] = (1.f - zg) * ng + zg * hp;
    }
  }
  __syncthreads();

  // lane attention scores over concat(lane_features, h)
  if (tid < MAX_L) {
    float s = lattn_b[0];
    for (int d = 0; d < LDIM; ++d) s += lattn_w[d] * (float)s_lf[tid * 32 + d];
    for (int k = 0; k < HID; ++k)  s += lattn_w[LDIM + k] * s_h[tid * HID + k];
    s_sc[tid] = s;
  }
  __syncthreads();
  if (tid == 0) {
    float mx = -1e30f;
    for (int l = 0; l < MAX_L; ++l) mx = fmaxf(mx, s_sc[l]);
    float den = 0.f;
    for (int l = 0; l < MAX_L; ++l) den += __expf(s_sc[l] - mx);
    s_red[0] = mx; s_red[1] = den;
  }
  __syncthreads();
  {
    float mx = s_red[0], den = s_red[1], acc = 0.f;
    for (int l = 0; l < MAX_L; ++l)
      acc += (__expf(s_sc[l] - mx) / den) * s_h[l * HID + tid];
    agg[(size_t)agent * HID + tid]  = acc;
    aggh[(size_t)agent * HID + tid] = (_Float16)acc;
  }
}

// ---------------------------------------------------------------------------
// K2: hg = agg @ gfc^T + b ; a_i = hg @ ga1A^T ; a_j = hg @ ga1B^T
//     one 16-row tile per block (8 batches x 16 tiles = 128 blocks)
// ---------------------------------------------------------------------------
__global__ __launch_bounds__(128) void gat_linear_kernel(
    const _Float16* __restrict__ aggh, const _Float16* __restrict__ gfch,
    const float* __restrict__ gfc_b, const _Float16* __restrict__ ga1A,
    const _Float16* __restrict__ ga1B, float* __restrict__ ai,
    float* __restrict__ aj, _Float16* __restrict__ hgT) {
  __shared__ _Float16 s_hg[16 * HID];
  const int tid = threadIdx.x, lane = tid & 31, wave = tid >> 5;
  const int b = blockIdx.x >> 4, i0 = (blockIdx.x & 15) * 16;
  const _Float16* A = aggh + ((size_t)b * N_AGENTS + i0) * HID;

  __builtin_prefetch((const char*)gfch + tid * 256, 0, 1);

  for (int t = wave; t < 8; t += 4) {
    v8f c = {};
#pragma unroll
    for (int k0 = 0; k0 < HID; k0 += 32)
      c = wmma_f16(load_frag_a_h(A, HID, k0, lane),
                   load_frag_b_h(gfch, HID, t * 16, k0, lane), c);
    int col = t * 16 + (lane & 15), mh = ((lane >> 4) & 1) * 8;
    float bv = gfc_b[col];
#pragma unroll
    for (int r = 0; r < 8; ++r) {
      float v = c[r] + bv;
      s_hg[(mh + r) * HID + col] = (_Float16)v;
      // transposed f16 copy: B-operand for h_prime = aw @ hg
      hgT[(size_t)b * (HID * N_AGENTS) + (size_t)col * N_AGENTS + i0 + mh + r] =
          (_Float16)v;
    }
  }
  __syncthreads();

  for (int t = wave; t < 8; t += 4) {
    v8f ci = {}, cj = {};
#pragma unroll
    for (int k0 = 0; k0 < HID; k0 += 32) {
      v16h a = load_frag_a_h(s_hg, HID, k0, lane);
      ci = wmma_f16(a, load_frag_b_h(ga1A, HID, t * 16, k0, lane), ci);
      cj = wmma_f16(a, load_frag_b_h(ga1B, HID, t * 16, k0, lane), cj);
    }
    int col = t * 16 + (lane & 15), mh = ((lane >> 4) & 1) * 8;
#pragma unroll
    for (int r = 0; r < 8; ++r) {
      size_t idx = ((size_t)b * N_AGENTS + i0 + mh + r) * HID + col;
      ai[idx] = ci[r];
      aj[idx] = cj[r];
    }
  }
}

// ---------------------------------------------------------------------------
// K3: e[b,i,j] = sum_h relu(a_i + a_j + b1) * w2 + b2, mask, row softmax.
//     relu inside the contraction -> VALU stage.  One (b,i) row per block.
// ---------------------------------------------------------------------------
__global__ __launch_bounds__(256) void edge_softmax_kernel(
    const float* __restrict__ ai, const float* __restrict__ aj,
    const float* __restrict__ ga1_b, const float* __restrict__ ga2_w,
    const float* __restrict__ ga2_b, const float* __restrict__ adj,
    _Float16* __restrict__ awh) {
  __shared__ float s_ai[HID];
  __shared__ float s_g2[HID];
  __shared__ float s_red[256];
  const int tid = threadIdx.x;
  const int b = blockIdx.x >> 8, i = blockIdx.x & 255;

  if (tid < HID) {
    s_ai[tid] = ai[((size_t)b * N_AGENTS + i) * HID + tid] + ga1_b[tid];
    s_g2[tid] = ga2_w[tid];
  }
  __syncthreads();

  const int j = tid;
  const float* ajr = aj + ((size_t)b * N_AGENTS + j) * HID;
  float s = 0.f;
#pragma unroll 4
  for (int h = 0; h < HID; ++h)
    s += fmaxf(s_ai[h] + ajr[h], 0.f) * s_g2[h];
  float e = s + ga2_b[0];
  if (adj[i * N_AGENTS + j] == 0.f) e = -1e9f;

  s_red[tid] = e;
  __syncthreads();
  for (int off = 128; off > 0; off >>= 1) {
    if (tid < off) s_red[tid] = fmaxf(s_red[tid], s_red[tid + off]);
    __syncthreads();
  }
  float mx = s_red[0];
  __syncthreads();
  float ex = __expf(e - mx);
  s_red[tid] = ex;
  __syncthreads();
  for (int off = 128; off > 0; off >>= 1) {
    if (tid < off) s_red[tid] += s_red[tid + off];
    __syncthreads();
  }
  float den = s_red[0];
  awh[((size_t)b * N_AGENTS + i) * N_AGENTS + j] = (_Float16)(ex / den);
}

// ---------------------------------------------------------------------------
// K4: h' = aw @ hg ; comm = h' @ gout^T + b ; q = [agg, comm] @ fc2^T + b
//     one 16-row tile per block (128 blocks)
// ---------------------------------------------------------------------------
__global__ __launch_bounds__(128) void hprime_out_kernel(
    const _Float16* __restrict__ awh, const _Float16* __restrict__ hgT,
    const _Float16* __restrict__ gouth, const float* __restrict__ gout_b,
    const _Float16* __restrict__ aggh, const _Float16* __restrict__ fc2h,
    const float* __restrict__ fc2_b, float* __restrict__ q) {
  __shared__ _Float16 s_hp[16 * HID];
  __shared__ _Float16 s_cat[16 * 256];
  const int tid = threadIdx.x, lane = tid & 31, wave = tid >> 5;
  const int b = blockIdx.x >> 4, i0 = (blockIdx.x & 15) * 16;

  const _Float16* A  = awh + ((size_t)b * N_AGENTS + i0) * N_AGENTS; // 16x256
  const _Float16* Bh = hgT + (size_t)b * (HID * N_AGENTS);           // [128][256]

  // h_prime = aw @ hg : K = 256
  for (int t = wave; t < 8; t += 4) {
    v8f c = {};
#pragma unroll
    for (int k0 = 0; k0 < N_AGENTS; k0 += 32)
      c = wmma_f16(load_frag_a_h(A, N_AGENTS, k0, lane),
                   load_frag_b_h(Bh, N_AGENTS, t * 16, k0, lane), c);
    int col = t * 16 + (lane & 15), mh = ((lane >> 4) & 1) * 8;
#pragma unroll
    for (int r = 0; r < 8; ++r) s_hp[(mh + r) * HID + col] = (_Float16)c[r];
  }
  // left half of concat: aggregated
  for (int idx = tid; idx < 16 * HID; idx += 128) {
    int m = idx >> 7, k = idx & 127;
    s_cat[m * 256 + k] = aggh[((size_t)b * N_AGENTS + i0 + m) * HID + k];
  }
  __syncthreads();

  // comm = h_prime @ gout^T + gout_b -> right half of concat
  for (int t = wave; t < 8; t += 4) {
    v8f c = {};
#pragma unroll
    for (int k0 = 0; k0 < HID; k0 += 32)
      c = wmma_f16(load_frag_a_h(s_hp, HID, k0, lane),
                   load_frag_b_h(gouth, HID, t * 16, k0, lane), c);
    int col = t * 16 + (lane & 15), mh = ((lane >> 4) & 1) * 8;
    float bv = gout_b[col];
#pragma unroll
    for (int r = 0; r < 8; ++r)
      s_cat[(mh + r) * 256 + HID + col] = (_Float16)(c[r] + bv);
  }
  __syncthreads();

  // q = cat @ fc2^T + fc2_b  (N padded to 16; only cols 0..7 stored)
  if (wave == 0) {
    v8f c = {};
#pragma unroll
    for (int k0 = 0; k0 < 256; k0 += 32)
      c = wmma_f16(load_frag_a_h(s_cat, 256, k0, lane),
                   load_frag_b_h(fc2h, 256, 0, k0, lane), c);
    int n = lane & 15, mh = ((lane >> 4) & 1) * 8;
    if (n < N_ACT) {
      float bv = fc2_b[n];
#pragma unroll
      for (int r = 0; r < 8; ++r)
        q[((size_t)b * N_AGENTS + i0 + mh + r) * N_ACT + n] = c[r] + bv;
    }
  }
}

// ---------------------------------------------------------------------------
extern "C" void kernel_launch(void* const* d_in, const int* in_sizes, int n_in,
                              void* d_out, int out_size, void* d_ws,
                              size_t ws_size, hipStream_t stream) {
  (void)in_sizes; (void)n_in; (void)out_size; (void)ws_size;
  const float* lf      = (const float*)d_in[0];
  const float* hin     = (const float*)d_in[1];
  const float* adj     = (const float*)d_in[2];
  const float* fc1_w   = (const float*)d_in[3];
  const float* fc1_b   = (const float*)d_in[4];
  const float* w_ih    = (const float*)d_in[5];
  const float* w_hh    = (const float*)d_in[6];
  const float* b_ih    = (const float*)d_in[7];
  const float* b_hh    = (const float*)d_in[8];
  const float* lattn_w = (const float*)d_in[9];
  const float* lattn_b = (const float*)d_in[10];
  const float* gfc_w   = (const float*)d_in[11];
  const float* gfc_b   = (const float*)d_in[12];
  const float* ga1_w   = (const float*)d_in[13];
  const float* ga1_b   = (const float*)d_in[14];
  const float* ga2_w   = (const float*)d_in[15];
  const float* ga2_b   = (const float*)d_in[16];
  const float* gout_w  = (const float*)d_in[17];
  const float* gout_b  = (const float*)d_in[18];
  const float* fc2_w   = (const float*)d_in[19];
  const float* fc2_b   = (const float*)d_in[20];

  char* ws = (char*)d_ws;
  _Float16* hb = (_Float16*)ws;                                   // 344064 B
  float*    fb = (float*)(ws + 344064);                           // 3 MiB
  _Float16* h2 = (_Float16*)(ws + 344064 + 3145728);              // 2 MiB
  float*    agg  = fb + F_AGG;
  float*    ai   = fb + F_AI;
  float*    aj   = fb + F_AJ;
  _Float16* aggh = h2 + H2_AGGH;
  _Float16* hgT  = h2 + H2_HGT;
  _Float16* awh  = h2 + H2_AWH;

  convert_weights_kernel<<<(HALF_TOT + 255) / 256, 256, 0, stream>>>(
      fc1_w, w_ih, w_hh, gfc_w, ga1_w, gout_w, fc2_w, hb);

  lane_agent_kernel<<<BN, 128, 0, stream>>>(
      lf, hin, hb + OFF_FC1H, fc1_b, hb + OFF_WIH, hb + OFF_WHH, b_ih, b_hh,
      lattn_w, lattn_b, agg, aggh);

  gat_linear_kernel<<<BS * 16, 128, 0, stream>>>(
      aggh, hb + OFF_GFC, gfc_b, hb + OFF_GA1A, hb + OFF_GA1B, ai, aj, hgT);

  edge_softmax_kernel<<<BS * N_AGENTS, 256, 0, stream>>>(
      ai, aj, ga1_b, ga2_w, ga2_b, adj, awh);

  hprime_out_kernel<<<BS * 16, 128, 0, stream>>>(
      awh, hgT, hb + OFF_GOUT, gout_b, aggh, hb + OFF_FC2, fc2_b,
      (float*)d_out);
}
